// GINE_Allocation_Predictor_31421980738093
// MI455X (gfx1250) — matrix operand order repaired and checked by reading, not compiled, not run on hardware
//
#include <hip/hip_runtime.h>
#include <hip/hip_bf16.h>
#include <math.h>

// ---------------------------------------------------------------------------
// GINE allocation predictor, MI455X (gfx1250) version.
//   - edge phases: memory/atomic bound -> f16 hidden state to halve gather
//     traffic, fp32 global atomics for the scatter-add.
//   - node phases: dense [tile16 x K] x [K x 64] GEMMs via
//     v_wmma_f32_16x16x32_f16 (wave32, 16x16 tiles, f32 accumulate).
// ---------------------------------------------------------------------------

typedef __attribute__((ext_vector_type(16))) _Float16 v16h;
typedef __attribute__((ext_vector_type(8)))  float    v8f;

#define NN   50000
#define EE   1600000
#define GGR  512
#define FIN  9
#define HH   64

// ---- WMMA helpers (CDNA5 layouts from CDNA5 ISA) ----------------

static __device__ __forceinline__ v8f wmma_f16(v16h a, v16h b, v8f c) {
  // (neg_a, A, neg_b, B, c_mod, C, reuse_a, reuse_b)
  return __builtin_amdgcn_wmma_f32_16x16x32_f16(false, a, false, b,
                                                (short)0, c, false, false);
}

// A fragment: 16x32 f16, row-major source in LDS with 'stride' f16 per row.
// lane l<16 -> M=l, K in {0..7,16..23}; lane l>=16 -> M=l-16, K in {8..15,24..31}
static __device__ __forceinline__ v16h load_a_frag(const _Float16* lds, int row0,
                                                   int stride, int koff) {
  int lane = threadIdx.x & 31;
  const _Float16* p = lds + (row0 + (lane & 15)) * stride;
  int khalf = (lane >> 4) << 3;                 // 0 or 8
  v16h a;
#pragma unroll
  for (int v = 0; v < 8; ++v) {
    int kb = koff + ((v & 4) << 2) + khalf + ((v & 3) << 1);
    a[2 * v]     = p[kb];
    a[2 * v + 1] = p[kb + 1];
  }
  return a;
}

// B fragment: 32x16 (KxN) with B[k][n] = W[n][k]; W stored row-major in LDS
// (stride f16 per output row). lanes 0-15: N=lane, K {0..7,16..23}; lanes 16-31: K {8..15,24..31}.
static __device__ __forceinline__ v16h load_b_frag(const _Float16* wlds, int n0,
                                                   int stride, int koff) {
  int lane = threadIdx.x & 31;
  const _Float16* p = wlds + (n0 + (lane & 15)) * stride;
  int kh = (lane >> 4) << 3;                    // 0 or 8
  v16h b;
#pragma unroll
  for (int v = 0; v < 8; ++v) {
    int k = koff + ((v & 4) << 2) + kh + ((v & 3) << 1);
    b[2 * v]     = p[k];
    b[2 * v + 1] = p[k + 1];
  }
  return b;
}

// ---- misc kernels ----------------------------------------------------------

__global__ void k_zero(float* p, int n) {
  int i = blockIdx.x * blockDim.x + threadIdx.x;
  int strd = gridDim.x * blockDim.x;
  for (; i < n; i += strd) p[i] = 0.0f;
}

// conv1 edge phase: m = relu(x[src] + edge_attr*e1_w + e1_b); scatter-add to dst
__global__ void k_edges1(const float* __restrict__ x, const int* __restrict__ ei,
                         const float* __restrict__ ea, const float* __restrict__ e1w,
                         const float* __restrict__ e1b, float* __restrict__ aggr1) {
  int e = blockIdx.x * blockDim.x + threadIdx.x;
  if (e >= EE) return;
  int s = ei[e];
  int d = ei[EE + e];
  float a = ea[e];
#pragma unroll
  for (int j = 0; j < FIN; ++j) {
    float m = x[s * FIN + j] + a * e1w[j] + e1b[j];
    m = m > 0.0f ? m : 0.0f;
    atomicAdd(&aggr1[d * FIN + j], m);
  }
}

// conv1 node phase: h = relu(relu((x+aggr) @ w1a.T + b1a) @ w1b.T + b1b), f16 out
__global__ void __launch_bounds__(256)
k_nodes1(const float* __restrict__ x, const float* __restrict__ aggr1,
         const float* __restrict__ w1a, const float* __restrict__ b1a,
         const float* __restrict__ w1b, const float* __restrict__ b1b,
         _Float16* __restrict__ hOut) {
  __shared__ _Float16 sW1a[64 * 32];   // K padded 9 -> 32 with zeros
  __shared__ float    sB1a[64];
  __shared__ _Float16 sW1b[64 * 64];
  __shared__ float    sB1b[64];
  __shared__ _Float16 sAin[128 * 32];  // (x + aggr), K padded to 32
  __shared__ _Float16 sH1[128 * 64];

  int tid = threadIdx.x;
  int nodeBase = blockIdx.x * 128;

  for (int i = tid; i < 64 * 32; i += 256) {
    int r = i >> 5, k = i & 31;
    sW1a[i] = (k < FIN) ? (_Float16)w1a[r * FIN + k] : (_Float16)0.0f;
  }
  for (int i = tid; i < 64 * 64; i += 256) sW1b[i] = (_Float16)w1b[i];
  if (tid < 64) { sB1a[tid] = b1a[tid]; sB1b[tid] = b1b[tid]; }

  for (int i = tid; i < 128 * 32; i += 256) {
    int r = i >> 5, k = i & 31;
    int node = nodeBase + r;
    float v = 0.0f;
    if (node < NN && k < FIN) v = x[node * FIN + k] + aggr1[node * FIN + k];
    sAin[i] = (_Float16)v;
  }
  __syncthreads();

  int wid = tid >> 5;
  int r0 = wid << 4;
  int lane = tid & 31;
  int col16 = lane & 15;
  int rb = (lane >> 4) << 3;

  // layer 1: [16 x 32pad] x [32 x 64]
  {
    v16h a = load_a_frag(sAin, r0, 32, 0);
#pragma unroll
    for (int nt = 0; nt < 4; ++nt) {
      v16h b = load_b_frag(sW1a, nt * 16, 32, 0);
      v8f c = {};
      c = wmma_f16(a, b, c);
      int col = nt * 16 + col16;
      float bias = sB1a[col];
#pragma unroll
      for (int r = 0; r < 8; ++r) {
        float v = c[r] + bias;
        sH1[(r0 + rb + r) * 64 + col] = (_Float16)(v > 0.0f ? v : 0.0f);
      }
    }
  }
  __syncthreads();

  // layer 2: [16 x 64] x [64 x 64], write f16 hidden state to global
#pragma unroll
  for (int nt = 0; nt < 4; ++nt) {
    v8f c = {};
#pragma unroll
    for (int ks = 0; ks < 2; ++ks) {
      v16h a = load_a_frag(sH1, r0, 64, ks * 32);
      v16h b = load_b_frag(sW1b, nt * 16, 64, ks * 32);
      c = wmma_f16(a, b, c);
    }
    int col = nt * 16 + col16;
    float bias = sB1b[col];
#pragma unroll
    for (int r = 0; r < 8; ++r) {
      int node = nodeBase + r0 + rb + r;
      if (node < NN) {
        float v = c[r] + bias;
        hOut[node * 64 + col] = (_Float16)(v > 0.0f ? v : 0.0f);
      }
    }
  }
}

// conv2 edge phase: one thread per (edge, channel); f16 gather, fp32 atomics
__global__ void k_edges2(const _Float16* __restrict__ h, const int* __restrict__ ei,
                         const float* __restrict__ ea, const float* __restrict__ e2w,
                         const float* __restrict__ e2b, float* __restrict__ aggr2) {
  long long idx = (long long)blockIdx.x * blockDim.x + threadIdx.x;
  if (idx >= (long long)EE * 64) return;
  int e = (int)(idx >> 6);
  int n = (int)(idx & 63);
  int s = ei[e];
  int d = ei[EE + e];
  float m = (float)h[s * 64 + n] + ea[e] * e2w[n] + e2b[n];
  m = m > 0.0f ? m : 0.0f;
  atomicAdd(&aggr2[d * 64 + n], m);
}

// conv2 node phase + readout head + per-graph expense accumulation
__global__ void __launch_bounds__(256)
k_nodes2(const _Float16* __restrict__ h, const float* __restrict__ aggr2,
         const float* __restrict__ w2a, const float* __restrict__ b2a,
         const float* __restrict__ w2b, const float* __restrict__ b2b,
         const float* __restrict__ wr1, const float* __restrict__ br1,
         const float* __restrict__ wr2, const float* __restrict__ br2,
         const int* __restrict__ term, const float* __restrict__ ccost,
         const int* __restrict__ batch, float* __restrict__ pi,
         float* __restrict__ totexp) {
  __shared__ _Float16 sW2a[64 * 64];
  __shared__ _Float16 sW2b[64 * 64];
  __shared__ _Float16 sWr1[32 * 64];
  __shared__ float sB2a[64], sB2b[64], sBr1[32], sWr2[32];
  __shared__ float sBr2;
  __shared__ _Float16 sA[128 * 64];
  __shared__ _Float16 sH[128 * 64];
  __shared__ _Float16 sH3[128 * 32];

  int tid = threadIdx.x;
  int nodeBase = blockIdx.x * 128;

  for (int i = tid; i < 64 * 64; i += 256) {
    sW2a[i] = (_Float16)w2a[i];
    sW2b[i] = (_Float16)w2b[i];
  }
  for (int i = tid; i < 32 * 64; i += 256) sWr1[i] = (_Float16)wr1[i];
  if (tid < 64) { sB2a[tid] = b2a[tid]; sB2b[tid] = b2b[tid]; }
  if (tid < 32) { sBr1[tid] = br1[tid]; sWr2[tid] = wr2[tid]; }
  if (tid == 0) sBr2 = br2[0];

  for (int i = tid; i < 128 * 64; i += 256) {
    int r = i >> 6, k = i & 63;
    int node = nodeBase + r;
    float v = 0.0f;
    if (node < NN) v = (float)h[node * 64 + k] + aggr2[node * 64 + k];
    sA[i] = (_Float16)v;
  }
  __syncthreads();

  int wid = tid >> 5;
  int r0 = wid << 4;
  int lane = tid & 31;
  int col16 = lane & 15;
  int rb = (lane >> 4) << 3;

  // GEMM w2a: sA -> sH
#pragma unroll
  for (int nt = 0; nt < 4; ++nt) {
    v8f c = {};
#pragma unroll
    for (int ks = 0; ks < 2; ++ks) {
      v16h a = load_a_frag(sA, r0, 64, ks * 32);
      v16h b = load_b_frag(sW2a, nt * 16, 64, ks * 32);
      c = wmma_f16(a, b, c);
    }
    int col = nt * 16 + col16;
    float bias = sB2a[col];
#pragma unroll
    for (int r = 0; r < 8; ++r) {
      float v = c[r] + bias;
      sH[(r0 + rb + r) * 64 + col] = (_Float16)(v > 0.0f ? v : 0.0f);
    }
  }
  __syncthreads();

  // GEMM w2b: sH -> sA (reuse)
#pragma unroll
  for (int nt = 0; nt < 4; ++nt) {
    v8f c = {};
#pragma unroll
    for (int ks = 0; ks < 2; ++ks) {
      v16h a = load_a_frag(sH, r0, 64, ks * 32);
      v16h b = load_b_frag(sW2b, nt * 16, 64, ks * 32);
      c = wmma_f16(a, b, c);
    }
    int col = nt * 16 + col16;
    float bias = sB2b[col];
#pragma unroll
    for (int r = 0; r < 8; ++r) {
      float v = c[r] + bias;
      sA[(r0 + rb + r) * 64 + col] = (_Float16)(v > 0.0f ? v : 0.0f);
    }
  }
  __syncthreads();

  // GEMM wr1 (64 -> 32): sA -> sH3
#pragma unroll
  for (int nt = 0; nt < 2; ++nt) {
    v8f c = {};
#pragma unroll
    for (int ks = 0; ks < 2; ++ks) {
      v16h a = load_a_frag(sA, r0, 64, ks * 32);
      v16h b = load_b_frag(sWr1, nt * 16, 64, ks * 32);
      c = wmma_f16(a, b, c);
    }
    int col = nt * 16 + col16;
    float bias = sBr1[col];
#pragma unroll
    for (int r = 0; r < 8; ++r) {
      float v = c[r] + bias;
      sH3[(r0 + rb + r) * 32 + col] = (_Float16)(v > 0.0f ? v : 0.0f);
    }
  }
  __syncthreads();

  // readout head: sigmoid(dot(h3, wr2) + br2), mask, expense atomics
  if (tid < 128) {
    int node = nodeBase + tid;
    if (node < NN) {
      float s = sBr2;
#pragma unroll
      for (int k = 0; k < 32; ++k) s += (float)sH3[tid * 32 + k] * sWr2[k];
      float p = 1.0f / (1.0f + expf(-s));
      p *= (1.0f - (float)term[node]);
      pi[node] = p;
      float ex = p * ccost[node];
      atomicAdd(&totexp[batch[node]], ex);
    }
  }
}

__global__ void k_final(const float* __restrict__ pi, const int* __restrict__ batch,
                        const float* __restrict__ Bt, const float* __restrict__ totexp,
                        float* __restrict__ out) {
  int i = blockIdx.x * blockDim.x + threadIdx.x;
  if (i >= NN) return;
  int b = batch[i];
  float ratio = Bt[b] / (totexp[b] + 1e-12f);
  ratio = ratio < 1.0f ? ratio : 1.0f;
  out[i] = pi[i] * ratio;
}

// ---- launch ----------------------------------------------------------------

extern "C" void kernel_launch(void* const* d_in, const int* in_sizes, int n_in,
                              void* d_out, int out_size, void* d_ws, size_t ws_size,
                              hipStream_t stream) {
  const float* x    = (const float*)d_in[0];
  const int*   ei   = (const int*)d_in[1];
  const float* ea   = (const float*)d_in[2];
  const int*   bat  = (const int*)d_in[3];
  const float* Bt   = (const float*)d_in[4];
  const int*   term = (const int*)d_in[5];
  const float* cc   = (const float*)d_in[6];
  const float* e1w  = (const float*)d_in[7];
  const float* e1b  = (const float*)d_in[8];
  const float* w1a  = (const float*)d_in[9];
  const float* b1a  = (const float*)d_in[10];
  const float* w1b  = (const float*)d_in[11];
  const float* b1b  = (const float*)d_in[12];
  const float* e2w  = (const float*)d_in[13];
  const float* e2b  = (const float*)d_in[14];
  const float* w2a  = (const float*)d_in[15];
  const float* b2a  = (const float*)d_in[16];
  const float* w2b  = (const float*)d_in[17];
  const float* b2b  = (const float*)d_in[18];
  const float* wr1  = (const float*)d_in[19];
  const float* br1  = (const float*)d_in[20];
  const float* wr2  = (const float*)d_in[21];
  const float* br2  = (const float*)d_in[22];
  float* out = (float*)d_out;

  // workspace layout (bytes, 16-aligned)
  char* ws = (char*)d_ws;
  float*    aggr1  = (float*)(ws);                               // N*9  f32
  float*    aggr2  = (float*)(ws + (size_t)NN * FIN * 4);        // N*64 f32
  float*    totexp = (float*)(ws + (size_t)NN * FIN * 4
                                 + (size_t)NN * HH * 4);         // G    f32
  float*    pi     = (float*)((char*)totexp + (size_t)GGR * 4);  // N    f32
  _Float16* hbuf   = (_Float16*)((char*)pi + (size_t)NN * 4);    // N*64 f16

  // 1) zero accumulators (aggr1 | aggr2 | totexp are contiguous)
  int zeroN = NN * FIN + NN * HH + GGR;
  k_zero<<<2048, 256, 0, stream>>>(aggr1, zeroN);

  // 2) conv1 edges
  k_edges1<<<(EE + 255) / 256, 256, 0, stream>>>(x, ei, ea, e1w, e1b, aggr1);

  // 3) conv1 node GEMMs (WMMA) -> f16 hidden state
  k_nodes1<<<(NN + 127) / 128, 256, 0, stream>>>(x, aggr1, w1a, b1a, w1b, b1b, hbuf);

  // 4) conv2 edges (f16 gather, fp32 atomics)
  long long w2 = (long long)EE * 64;
  k_edges2<<<(unsigned)((w2 + 255) / 256), 256, 0, stream>>>(hbuf, ei, ea, e2w, e2b, aggr2);

  // 5) conv2 node GEMMs + readout head (WMMA)
  k_nodes2<<<(NN + 127) / 128, 256, 0, stream>>>(hbuf, aggr2, w2a, b2a, w2b, b2b,
                                                 wr1, br1, wr2, br2,
                                                 term, cc, bat, pi, totexp);

  // 6) budget ratio finalize
  k_final<<<(NN + 255) / 256, 256, 0, stream>>>(pi, bat, Bt, totexp, out);
}
